// CompressAttn_49598282334700
// MI455X (gfx1250) — matrix operand order repaired and checked by reading, not compiled, not run on hardware
//
#include <hip/hip_runtime.h>

typedef __attribute__((ext_vector_type(16))) __bf16 v16bf;
typedef __attribute__((ext_vector_type(8)))  float  v8f;

#define B_    2
#define N_    4096
#define QH_   32
#define KH_   2
#define D_    128
#define VD_   128
#define KSZ_  32
#define STRIDE_ 16
#define M_    255
#define MPAD_ 256
#define NEGV  (-1e30f)
#define SCALE 0.08838834764831845f   // 1/sqrt(128)

// ---------------- compression: kc[bh][m][d] (bf16), vct[bh][vd][m] (bf16) ----------------
__global__ __launch_bounds__(128) void nsa_compress(
    const float* __restrict__ k,  const float* __restrict__ v,
    const float* __restrict__ wk, const float* __restrict__ pek,
    const float* __restrict__ wv, const float* __restrict__ pev,
    __bf16* __restrict__ kc, __bf16* __restrict__ vct)
{
  int m = blockIdx.x, h = blockIdx.y, b = blockIdx.z, d = threadIdx.x;
  int bh = b * KH_ + h;
  if (m >= M_) {            // zero pad row so attention can load it unconditionally
    kc[((size_t)bh * MPAD_ + m) * D_ + d] = (__bf16)0.0f;
    vct[((size_t)bh * VD_ + d) * MPAD_ + m] = (__bf16)0.0f;
    return;
  }
  float aK = 0.0f, aV = 0.0f;
  #pragma unroll 4
  for (int j = 0; j < KSZ_; ++j) {
    int n = m * STRIDE_ + j;
    size_t base = ((size_t)(b * N_ + n) * KH_ + h) * D_ + d;
    aK += wk[j] * (k[base] + pek[j * D_  + d]);
    aV += wv[j] * (v[base] + pev[j * VD_ + d]);
  }
  kc[((size_t)bh * MPAD_ + m) * D_ + d] = (__bf16)aK;
  vct[((size_t)bh * VD_ + d) * MPAD_ + m] = (__bf16)aV;
}

// load 16 contiguous bf16 (32B aligned) as a WMMA operand vector
__device__ inline v16bf loadB16(const __bf16* p) {
  union { uint4 u[2]; v16bf v; } c;
  c.u[0] = *(const uint4*)p;
  c.u[1] = *(const uint4*)(p + 8);
  return c.v;
}

// ---------------- flash attention over compressed keys ----------------
// block = 256 threads = 8 waves; wave w handles q-head blockIdx.y*8+w, query tile blockIdx.x
__global__ __launch_bounds__(256) void nsa_attn(
    const float* __restrict__ q, const __bf16* __restrict__ kc,
    const __bf16* __restrict__ vct, float* __restrict__ out)
{
  __shared__ float sP[8 * 16 * 34];   // per-wave 16x32 P scratch, stride 34 to dodge banks

  const int wave = threadIdx.x >> 5, lane = threadIdx.x & 31;
  const int hi = lane >> 4, lm = lane & 15;
  const int t  = blockIdx.x;                  // query tile (16 rows)
  const int hq = blockIdx.y * 8 + wave;       // query head
  const int b  = blockIdx.z;
  const int hkv = hq >> 4;                    // GQA: G = 16
  const int bh  = b * KH_ + hkv;
  const int n_base = t * 16;

  // ---- load Q rows (n = n_base + lane%16) into bf16 A-layout, 4 chunks of K=32 ----
  const float* qrow = q + ((size_t)(b * N_ + n_base + lm) * QH_ + hq) * D_;
  v16bf qa[4];
  #pragma unroll
  for (int dc = 0; dc < 4; ++dc) {
    const float* p0 = qrow + dc * 32 + 16 * hi;  // K = 16*hi + 0..7
    const float* p1 = p0 + 8;                    // K = 16*hi + 8..15
    float4 x0 = *(const float4*)p0;
    float4 x1 = *(const float4*)(p0 + 4);
    float4 y0 = *(const float4*)p1;
    float4 y1 = *(const float4*)(p1 + 4);
    v16bf a;
    a[0]=(__bf16)x0.x; a[1]=(__bf16)x0.y; a[2]=(__bf16)x0.z; a[3]=(__bf16)x0.w;
    a[4]=(__bf16)x1.x; a[5]=(__bf16)x1.y; a[6]=(__bf16)x1.z; a[7]=(__bf16)x1.w;
    a[8]=(__bf16)y0.x; a[9]=(__bf16)y0.y; a[10]=(__bf16)y0.z; a[11]=(__bf16)y0.w;
    a[12]=(__bf16)y1.x; a[13]=(__bf16)y1.y; a[14]=(__bf16)y1.z; a[15]=(__bf16)y1.w;
    qa[dc] = a;
  }

  v8f o[8];
  float m_i[8], l_i[8];
  #pragma unroll
  for (int i = 0; i < 8; ++i) {
    o[i] = (v8f){0,0,0,0,0,0,0,0};
    m_i[i] = -__builtin_inff();
    l_i[i] = 0.0f;
  }

  // valid compressed keys: m <= t-1  ->  chunks of 32 keys
  const int nchunks = (t + 31) >> 5;            // <= 8
  float* myP = sP + wave * 544;

  for (int c = 0; c < nchunks; ++c) {
    // ---- S = Q K^T for 2 key tiles (keys 32c..32c+31) ----
    v8f s0 = (v8f){0,0,0,0,0,0,0,0};
    v8f s1 = (v8f){0,0,0,0,0,0,0,0};
    const __bf16* kb = kc + ((size_t)bh * MPAD_) * D_;
    #pragma unroll
    for (int dc = 0; dc < 4; ++dc) {
      v16bf b0 = loadB16(kb + (size_t)(32 * c + lm) * D_ + dc * 32 + 16 * hi);
      s0 = __builtin_amdgcn_wmma_f32_16x16x32_bf16(false, qa[dc], false, b0,
                                                   (short)0, s0, false, false);
    }
    #pragma unroll
    for (int dc = 0; dc < 4; ++dc) {
      v16bf b1 = loadB16(kb + (size_t)(32 * c + 16 + lm) * D_ + dc * 32 + 16 * hi);
      s1 = __builtin_amdgcn_wmma_f32_16x16x32_bf16(false, qa[dc], false, b1,
                                                   (short)0, s1, false, false);
    }

    // ---- scale, causal mask, online softmax (rows r+8*hi, cols = lane%16) ----
    float al[8], p0v[8], p1v[8];
    const int m0 = 32 * c + lm, m1 = m0 + 16;
    #pragma unroll
    for (int r = 0; r < 8; ++r) {
      int nrow = n_base + r + 8 * hi;
      float a0 = (nrow >= m0 * 16 + 31) ? s0[r] * SCALE : NEGV;
      float a1 = (nrow >= m1 * 16 + 31) ? s1[r] * SCALE : NEGV;
      float mx = fmaxf(a0, a1);
      mx = fmaxf(mx, __shfl_xor(mx, 1));
      mx = fmaxf(mx, __shfl_xor(mx, 2));
      mx = fmaxf(mx, __shfl_xor(mx, 4));
      mx = fmaxf(mx, __shfl_xor(mx, 8));
      float mn = fmaxf(m_i[r], mx);
      float alpha = __expf(m_i[r] - mn);
      float e0 = __expf(a0 - mn), e1 = __expf(a1 - mn);
      float rs = e0 + e1;
      rs += __shfl_xor(rs, 1);
      rs += __shfl_xor(rs, 2);
      rs += __shfl_xor(rs, 4);
      rs += __shfl_xor(rs, 8);
      l_i[r] = l_i[r] * alpha + rs;
      m_i[r] = mn;
      al[r] = alpha;
      p0v[r] = e0; p1v[r] = e1;
    }

    // ---- C-layout -> A-layout for P via per-wave LDS scratch ----
    #pragma unroll
    for (int r = 0; r < 8; ++r) {
      int rr = r + 8 * hi;
      myP[rr * 34 + lm]      = p0v[r];
      myP[rr * 34 + 16 + lm] = p1v[r];
    }
    asm volatile("s_wait_dscnt 0" ::: "memory");
    v16bf pa;
    {
      const float* pr = myP + lm * 34;
      #pragma unroll
      for (int j = 0; j < 4; ++j) {
        float2 e = *(const float2*)(pr + 16 * hi + 2 * j);
        pa[2 * j]     = (__bf16)e.x;
        pa[2 * j + 1] = (__bf16)e.y;
      }
      #pragma unroll
      for (int j = 0; j < 4; ++j) {
        float2 e = *(const float2*)(pr + 16 * hi + 8 + 2 * j);
        pa[8 + 2 * j] = (__bf16)e.x;
        pa[9 + 2 * j] = (__bf16)e.y;
      }
    }

    // ---- rescale O, then O += P V ----
    #pragma unroll
    for (int vt = 0; vt < 8; ++vt) {
      #pragma unroll
      for (int r = 0; r < 8; ++r) o[vt][r] *= al[r];
    }
    const __bf16* vb_base = vct + (size_t)(bh * VD_) * MPAD_ + 32 * c + 16 * hi;
    #pragma unroll
    for (int vt = 0; vt < 8; ++vt) {
      v16bf vb = loadB16(vb_base + (size_t)(16 * vt + lm) * MPAD_);
      o[vt] = __builtin_amdgcn_wmma_f32_16x16x32_bf16(false, pa, false, vb,
                                                      (short)0, o[vt], false, false);
    }
  }

  // ---- normalize and store; rows n < KSZ-1 are zeroed (reference `valid` mask) ----
  float inv[8];
  #pragma unroll
  for (int r = 0; r < 8; ++r) {
    int nrow = n_base + r + 8 * hi;
    inv[r] = (nrow >= KSZ_ - 1 && l_i[r] > 0.0f) ? 1.0f / l_i[r] : 0.0f;
  }
  #pragma unroll
  for (int vt = 0; vt < 8; ++vt) {
    #pragma unroll
    for (int r = 0; r < 8; ++r) {
      int nrow = n_base + r + 8 * hi;
      out[((size_t)(b * N_ + nrow) * QH_ + hq) * VD_ + 16 * vt + lm] = o[vt][r] * inv[r];
    }
  }
}

extern "C" void kernel_launch(void* const* d_in, const int* in_sizes, int n_in,
                              void* d_out, int out_size, void* d_ws, size_t ws_size,
                              hipStream_t stream) {
  const float* q   = (const float*)d_in[0];
  const float* k   = (const float*)d_in[1];
  const float* v   = (const float*)d_in[2];
  const float* wk  = (const float*)d_in[3];
  const float* pek = (const float*)d_in[4];
  const float* wv  = (const float*)d_in[5];
  const float* pev = (const float*)d_in[6];

  __bf16* kc  = (__bf16*)d_ws;                                   // [B*KH][MPAD][D]
  __bf16* vct = kc + (size_t)B_ * KH_ * MPAD_ * D_;              // [B*KH][VD][MPAD]

  nsa_compress<<<dim3(MPAD_, KH_, B_), 128, 0, stream>>>(k, v, wk, pek, wv, pev, kc, vct);
  nsa_attn<<<dim3(N_ / 16, QH_ / 8, B_), 256, 0, stream>>>(q, kc, vct, (float*)d_out);
}
